// LocalAttention_21586505630180
// MI455X (gfx1250) — matrix operand order repaired and checked by reading, not compiled, not run on hardware
//
#include <hip/hip_runtime.h>

// LocalAttention (7x7 window MHA) for gfx1250 — WMMA f32_16x16x32_f16 path.
// B=16, C=96, H=W=224 -> 16384 windows of L=49 tokens, 8 heads of d=12.

#define DIMC   96
#define HEADS  8
#define HD     12
#define WSZ    7
#define LTOK   49
#define LPAD   64
#define HIMG   224
#define WIMG   224
#define HWSZ   (HIMG * WIMG)

typedef __attribute__((ext_vector_type(16))) _Float16 v16h;
typedef __attribute__((ext_vector_type(8)))  float    v8f;

union HU {
    v16h v;
    _Float16 h[16];
    unsigned u[8];
};

__device__ __forceinline__ v8f wmma_f16(v16h a, v16h b, v8f c) {
    return __builtin_amdgcn_wmma_f32_16x16x32_f16(
        /*neg_a=*/false, a, /*neg_b=*/false, b,
        /*c_mod=*/(short)0, c, /*reuse_a=*/false, /*reuse_b=*/false);
}

// A fragment 16x32: lane l(0..15)=row M=l (+hi selects K halves).
// VGPR v holds K = 2v + 16*hi (pairs). Source row-major in LDS.
__device__ __forceinline__ v16h frag_a_lds(const _Float16* base, int ld,
                                           int m0, int k0, int kvalid, int lane) {
    int r = lane & 15, hi = lane >> 4;
    const _Float16* rp = base + (m0 + r) * ld + k0;
    HU a;
#pragma unroll
    for (int v = 0; v < 8; ++v) {
        int kb = 2 * v + (hi << 4);
        a.h[2 * v]     = (kb     < kvalid) ? rp[kb]     : (_Float16)0.0f;
        a.h[2 * v + 1] = (kb + 1 < kvalid) ? rp[kb + 1] : (_Float16)0.0f;
    }
    return a.v;
}

// B fragment 32x16 where B[k][n] = src[(n0+n)*ld + koff + k]  (i.e. B = S^T,
// contiguous K pairs along a row -> used for K^T in the score GEMM).
__device__ __forceinline__ v16h frag_bt_lds(const _Float16* base, int ld,
                                            int n0, int koff, int kvalid, int lane) {
    int n = lane & 15, hi = lane >> 4;
    const _Float16* rp = base + (n0 + n) * ld + koff + (hi << 4);
    HU b;
#pragma unroll
    for (int j = 0; j < 8; ++j) {
        int k = 2 * j + (hi << 4);
        b.h[2 * j]     = (k     < kvalid) ? rp[2 * j]     : (_Float16)0.0f;
        b.h[2 * j + 1] = (k + 1 < kvalid) ? rp[2 * j + 1] : (_Float16)0.0f;
    }
    return b.v;
}

// B fragment 32x16 where B[k][n] = src[(k0+k)*ld + noff + n] (V matrix).
__device__ __forceinline__ v16h frag_b_lds(const _Float16* base, int ld,
                                           int k0, int noff, int nvalid, int lane) {
    int n = lane & 15, hi = lane >> 4;
    bool nv = n < nvalid;
    HU b;
#pragma unroll
    for (int j = 0; j < 8; ++j) {
        int k = k0 + 2 * j + (hi << 4);
        b.h[2 * j]     = nv ? base[k * ld + noff + n]       : (_Float16)0.0f;
        b.h[2 * j + 1] = nv ? base[(k + 1) * ld + noff + n] : (_Float16)0.0f;
    }
    return b.v;
}

// B fragment from global pre-packed weights: u32[Kpairs][ldn], each u32 holds
// f16 pair (k, k+1) -> one coalesced 32-bit load per VGPR.
__device__ __forceinline__ v16h frag_b_gw(const unsigned* __restrict__ W, int ldn,
                                          int k0, int n0, int lane) {
    int n = lane & 15, hi = lane >> 4;
    const unsigned* p = W + (size_t)((k0 >> 1) + (hi << 3)) * ldn + n0 + n;
    HU b;
#pragma unroll
    for (int j = 0; j < 8; ++j) b.u[j] = p[j * ldn];
    return b.v;
}

// ---------------- weight packing (runs once per launch, tiny) -------------
__global__ void pack_weights(const float* __restrict__ w_in,
                             const float* __restrict__ w_out,
                             unsigned* __restrict__ ws) {
    const int NQ = 48 * 288;   // w_in^T packed: [k/2][n(0..287)]
    const int NO = 48 * 96;    // w_out^T packed: [k/2][n(0..95)]
    int i = blockIdx.x * blockDim.x + threadIdx.x;
    if (i < NQ) {
        int kk = i / 288, n = i - kk * 288;
        union { _Float16 h[2]; unsigned u; } p;
        p.h[0] = (_Float16)w_in[n * 96 + 2 * kk];
        p.h[1] = (_Float16)w_in[n * 96 + 2 * kk + 1];
        ws[i] = p.u;
    } else if (i < NQ + NO) {
        int j = i - NQ;
        int kk = j / 96, n = j - kk * 96;
        union { _Float16 h[2]; unsigned u; } p;
        p.h[0] = (_Float16)w_out[n * 96 + 2 * kk];
        p.h[1] = (_Float16)w_out[n * 96 + 2 * kk + 1];
        ws[NQ + j] = p.u;
    }
}

// ---------------- fused window attention ----------------------------------
__global__ __launch_bounds__(256) void win_attn_kernel(
    const float* __restrict__ x, const float* __restrict__ b_in,
    const float* __restrict__ b_out, const unsigned* __restrict__ Wqkv,
    const unsigned* __restrict__ Wout, float* __restrict__ out) {

    __shared__ _Float16 Xs[LPAD][104];          // X window, f16, rows 49..63 = 0
    __shared__ _Float16 QKVs[LPAD][296];        // packed qkv, f16
    __shared__ _Float16 Ps[HEADS][16][72];      // per-wave softmax tile
    __shared__ _Float16 Os[LPAD][104];          // attention output, f16

    const int tid  = threadIdx.x;
    const int lane = tid & 31;
    const int wave = tid >> 5;
    const int li   = lane & 15;
    const int hi   = lane >> 4;

    const int wg  = blockIdx.x;                 // 16384 windows
    const int bb  = wg >> 10;
    const int rem = wg & 1023;
    const int h0  = (rem >> 5) * WSZ;
    const int w0  = (rem & 31) * WSZ;
    const float* xb = x + (size_t)bb * DIMC * HWSZ;

    // ---- load X window (49 tokens x 96 channels), transpose NCHW -> [tok][c]
    for (int e = tid; e < LTOK * DIMC; e += 256) {
        int c = e / LTOK, t = e - c * LTOK;
        int th = t / WSZ, tw = t - th * WSZ;
        Xs[t][c] = (_Float16)xb[c * HWSZ + (h0 + th) * WIMG + (w0 + tw)];
    }
    for (int e = tid; e < (LPAD - LTOK) * DIMC; e += 256) {
        int r = e / DIMC, c = e - r * DIMC;
        Xs[LTOK + r][c] = (_Float16)0.0f;
    }
    __syncthreads();

    // ---- phase 1: QKV = X @ Win^T + b_in   (64 x 288, K=96)
    {
        const int m0 = (wave & 3) << 4;
        v16h a0 = frag_a_lds(&Xs[0][0], 104, m0, 0,  32, lane);
        v16h a1 = frag_a_lds(&Xs[0][0], 104, m0, 32, 32, lane);
        v16h a2 = frag_a_lds(&Xs[0][0], 104, m0, 64, 32, lane);
        for (int nt = (wave >> 2); nt < 18; nt += 2) {
            const int n0 = nt << 4;
            v8f acc = {};
            acc = wmma_f16(a0, frag_b_gw(Wqkv, 288, 0,  n0, lane), acc);
            acc = wmma_f16(a1, frag_b_gw(Wqkv, 288, 32, n0, lane), acc);
            acc = wmma_f16(a2, frag_b_gw(Wqkv, 288, 64, n0, lane), acc);
            const int col = n0 + li;
            const float bias = b_in[col];
#pragma unroll
            for (int v = 0; v < 8; ++v)
                QKVs[m0 + v + (hi << 3)][col] = (_Float16)(acc[v] + bias);
        }
    }
    __syncthreads();

    // ---- phase 2: per-head attention, wave w == head w
    {
        const int hd = wave;
        const int qo = hd * HD;
        const int ko = DIMC + hd * HD;
        const int vo = 2 * DIMC + hd * HD;
        const float scale = 0.2886751345948129f;   // 1/sqrt(12)

        // K^T as B fragments (K padded 12->32 with zeros), reused for all M.
        v16h bk0 = frag_bt_lds(&QKVs[0][0], 296, 0,  ko, HD, lane);
        v16h bk1 = frag_bt_lds(&QKVs[0][0], 296, 16, ko, HD, lane);
        v16h bk2 = frag_bt_lds(&QKVs[0][0], 296, 32, ko, HD, lane);
        v16h bk3 = frag_bt_lds(&QKVs[0][0], 296, 48, ko, HD, lane);
        // V as B fragments (token pad rows are killed by P==0).
        v16h bv0 = frag_b_lds(&QKVs[0][0], 296, 0,  vo, HD, lane);
        v16h bv1 = frag_b_lds(&QKVs[0][0], 296, 32, vo, HD, lane);

        for (int m = 0; m < 4; ++m) {
            const int m0 = m << 4;
            v16h aq = frag_a_lds(&QKVs[0][0], 296, m0, qo, HD, lane);
            v8f s0 = {}, s1 = {}, s2 = {}, s3 = {};
            s0 = wmma_f16(aq, bk0, s0);
            s1 = wmma_f16(aq, bk1, s1);
            s2 = wmma_f16(aq, bk2, s2);
            s3 = wmma_f16(aq, bk3, s3);

            // masked softmax over row (cols >= 49 -> -inf); rows live in
            // (vgpr v, lane-half) -> reduce across the 16-lane half.
#pragma unroll
            for (int v = 0; v < 8; ++v) {
                float e0 = s0[v], e1 = s1[v], e2 = s2[v];
                float e3 = (li == 0) ? s3[v] : -3.0e38f;   // tile 3: only col 48 valid
                float mx = fmaxf(fmaxf(e0, e1), fmaxf(e2, e3));
#pragma unroll
                for (int s = 8; s >= 1; s >>= 1) mx = fmaxf(mx, __shfl_xor(mx, s, 32));
                float p0 = __expf(scale * (e0 - mx));
                float p1 = __expf(scale * (e1 - mx));
                float p2 = __expf(scale * (e2 - mx));
                float p3 = (li == 0) ? __expf(scale * (e3 - mx)) : 0.0f;
                float sm = p0 + p1 + p2 + p3;
#pragma unroll
                for (int s = 8; s >= 1; s >>= 1) sm += __shfl_xor(sm, s, 32);
                float inv = 1.0f / sm;
                int r = v + (hi << 3);
                Ps[hd][r][li]      = (_Float16)(p0 * inv);
                Ps[hd][r][16 + li] = (_Float16)(p1 * inv);
                Ps[hd][r][32 + li] = (_Float16)(p2 * inv);
                Ps[hd][r][48 + li] = (_Float16)(p3 * inv);
            }
            // same-wave LDS RAW: hardware keeps DS ops in order; fence compiler.
            asm volatile("" ::: "memory");

            v16h ap0 = frag_a_lds(&Ps[hd][0][0], 72, 0, 0,  32, lane);
            v16h ap1 = frag_a_lds(&Ps[hd][0][0], 72, 0, 32, 32, lane);
            v8f accO = {};
            accO = wmma_f16(ap0, bv0, accO);
            accO = wmma_f16(ap1, bv1, accO);
            if (li < HD) {
#pragma unroll
                for (int v = 0; v < 8; ++v)
                    Os[m0 + v + (hi << 3)][qo + li] = (_Float16)accO[v];
            }
        }
    }
    __syncthreads();

    // ---- phase 3: out = O @ Wout^T + b_out, scatter back to NCHW
    {
        const int m0 = (wave & 3) << 4;
        v16h a0 = frag_a_lds(&Os[0][0], 104, m0, 0,  32, lane);
        v16h a1 = frag_a_lds(&Os[0][0], 104, m0, 32, 32, lane);
        v16h a2 = frag_a_lds(&Os[0][0], 104, m0, 64, 32, lane);
        for (int nt = (wave >> 2); nt < 6; nt += 2) {
            const int n0 = nt << 4;
            v8f acc = {};
            acc = wmma_f16(a0, frag_b_gw(Wout, 96, 0,  n0, lane), acc);
            acc = wmma_f16(a1, frag_b_gw(Wout, 96, 32, n0, lane), acc);
            acc = wmma_f16(a2, frag_b_gw(Wout, 96, 64, n0, lane), acc);
            const int c = n0 + li;
            const float bias = b_out[c];
            float* ob = out + (size_t)(bb * DIMC + c) * HWSZ;
#pragma unroll
            for (int v = 0; v < 8; ++v) {
                int t = m0 + v + (hi << 3);
                if (t < LTOK) {
                    int th = t / WSZ, tw = t - th * WSZ;
                    ob[(h0 + th) * WIMG + (w0 + tw)] = acc[v] + bias;
                }
            }
        }
    }
}

extern "C" void kernel_launch(void* const* d_in, const int* in_sizes, int n_in,
                              void* d_out, int out_size, void* d_ws, size_t ws_size,
                              hipStream_t stream) {
    const float* x     = (const float*)d_in[0];
    const float* w_in  = (const float*)d_in[1];
    const float* b_in  = (const float*)d_in[2];
    const float* w_out = (const float*)d_in[3];
    const float* b_out = (const float*)d_in[4];
    float* out = (float*)d_out;

    unsigned* ws = (unsigned*)d_ws;           // [0,13824): Wqkv, [13824,18432): Wout
    pack_weights<<<(18432 + 255) / 256, 256, 0, stream>>>(w_in, w_out, ws);
    win_attn_kernel<<<16384, 256, 0, stream>>>(x, b_in, b_out,
                                               ws, ws + 48 * 288, out);
}
